// Encoder_4183298146503
// MI455X (gfx1250) — matrix operand order repaired and checked by reading, not compiled, not run on hardware
//
#include <hip/hip_runtime.h>
#include <hip/hip_bf16.h>

// ---------------------------------------------------------------------------
// Transformer encoder layer for MI455X (gfx1250, wave32, WMMA bf16 f32-acc)
// N=10, S=1024, H=768, A=12, DH=64, F=3072
// Weights pre-packed as K-pair uints so every WMMA fragment loads as 2x b128.
// ---------------------------------------------------------------------------

typedef __bf16  bf16x16 __attribute__((ext_vector_type(16)));
typedef float   floatx8 __attribute__((ext_vector_type(8)));

union AFrag {                 // 16 bf16 = 8 VGPRs per lane
    bf16x16        v;
    unsigned int   u[8];
    unsigned short s[16];
};

__device__ __forceinline__ floatx8 zero8() {
    floatx8 z = {0.f,0.f,0.f,0.f,0.f,0.f,0.f,0.f};
    return z;
}

__device__ __forceinline__ unsigned short f2bf(float f) {
    unsigned int u = __float_as_uint(f);
    unsigned int r = u + 0x7FFFu + ((u >> 16) & 1u);   // round-nearest-even
    return (unsigned short)(r >> 16);
}

// ---------------------------------------------------------------------------
// Elementwise f32 -> bf16 convert (activation side, row-major)
// ---------------------------------------------------------------------------
__global__ __launch_bounds__(256) void cvt_bf16(const float* __restrict__ in,
                                                unsigned short* __restrict__ out,
                                                int n) {
    int i = blockIdx.x * 256 + threadIdx.x;
    if (i < n) out[i] = f2bf(in[i]);
}

// f32 W[K][N] -> pair-packed uint Bp[K/2][N]: lo=W[2kp][n], hi=W[2kp+1][n]
__global__ __launch_bounds__(256) void cvt_pack_pairs(const float* __restrict__ W,
                                                      unsigned int* __restrict__ Bp,
                                                      int K, int N) {
    int i = blockIdx.x * 256 + threadIdx.x;
    if (i >= (K / 2) * N) return;
    int kp = i / N, n = i % N;
    unsigned int lo = f2bf(W[(size_t)(2 * kp)     * N + n]);
    unsigned int hi = f2bf(W[(size_t)(2 * kp + 1) * N + n]);
    Bp[i] = lo | (hi << 16);
}

// WQ/WK/WV (A,H,DH) -> pair-packed Wqkv [H/2, 2304], cols a*64+d | +768 | +1536
__global__ __launch_bounds__(256) void pack_qkv_pairs(const float* __restrict__ WQ,
                                                      const float* __restrict__ WK,
                                                      const float* __restrict__ WV,
                                                      unsigned int* __restrict__ Wqkv) {
    int i = blockIdx.x * 256 + threadIdx.x;       // over 384*2304
    if (i >= 384 * 2304) return;
    int kp = i / 2304;
    int c  = i % 2304;
    int sel = c / 768, cc = c % 768, a = cc / 64, d = cc % 64;
    const float* W = (sel == 0) ? WQ : (sel == 1) ? WK : WV;
    unsigned int lo = f2bf(W[(size_t)a * 768 * 64 + (size_t)(2 * kp)     * 64 + d]);
    unsigned int hi = f2bf(W[(size_t)a * 768 * 64 + (size_t)(2 * kp + 1) * 64 + d]);
    Wqkv[i] = lo | (hi << 16);
}

// ---------------------------------------------------------------------------
// Tiled bf16 GEMM with WMMA 16x16x32. BM=BN=128, BK=64, 8 waves, wave 64x32.
// A: bf16 [M,K] row-major. Bp: pair-packed uint [K/2,N].
// LDS double-buffered; B tile stored fragment-major [col][kpair] (pad 36).
// ---------------------------------------------------------------------------
#define BM 128
#define BN 128
#define BK 64
#define BPAD 36   // 32 kpairs + 4 pad; 36*4 bytes keeps 16B alignment

template<bool BIAS_, bool GELU_, bool RES_, bool OUTF_, bool OUTB_>
__global__ __launch_bounds__(256)
void gemm_bf16_wmma(const unsigned short* __restrict__ A,
                    const unsigned int*  __restrict__ Bp,
                    const float* __restrict__ bias,
                    const float* __restrict__ residual,
                    float* __restrict__ outF,
                    unsigned short* __restrict__ outB,
                    int M, int N, int K) {
    __shared__ uint4        As4[2][BM * BK / 8];      // 2 x 16 KB, row-major
    __shared__ unsigned int BsP[2][BN * BPAD];        // 2 x 18 KB, [col][kpair]

    const int tid  = threadIdx.x;
    const int lane = tid & 31;
    const int wid  = tid >> 5;
    const int wm   = wid >> 2;          // 0..1
    const int wn   = wid & 3;           // 0..3
    const int l16  = lane & 15;
    const int hs   = (lane >> 4) & 1;

    const int bm0 = blockIdx.y * BM;
    const int bn0 = blockIdx.x * BN;

    floatx8 acc[4][2];
#pragma unroll
    for (int mt = 0; mt < 4; ++mt)
#pragma unroll
        for (int nt = 0; nt < 2; ++nt) acc[mt][nt] = zero8();

    auto stage = [&](int buf, int k0) {
        // A tile: 128 rows x 8 uint4
#pragma unroll
        for (int i = tid, it = 0; it < BM * BK / 8 / 256; i += 256, ++it) {
            int r = i >> 3, c = i & 7;
            As4[buf][i] = *((const uint4*)(A + (size_t)(bm0 + r) * K + k0) + c);
        }
        // B tile: 32 kpair-rows x 128 cols, transposed into [col][kpair]
        int kp0 = k0 >> 1;
#pragma unroll
        for (int i = tid, it = 0; it < 32 * BN / 256; i += 256, ++it) {
            int r = i >> 7, c = i & 127;
            BsP[buf][c * BPAD + r] = Bp[(size_t)(kp0 + r) * N + bn0 + c];
        }
    };

    stage(0, 0);
    __syncthreads();

    for (int k0 = 0, it = 0; k0 < K; k0 += BK, ++it) {
        const int buf = it & 1;
        if (k0 + BK < K) stage(buf ^ 1, k0 + BK);
        if (k0 + 2 * BK < K) {   // prefetch two tiles ahead -> global_prefetch
            __builtin_prefetch(A + (size_t)(bm0 + (tid >> 1)) * K + k0 + 2 * BK, 0, 1);
            __builtin_prefetch(Bp + (size_t)((k0 + 2 * BK) >> 1) * N + bn0 + tid, 0, 1);
        }
        const unsigned int* Asu = (const unsigned int*)As4[buf];
        const unsigned int* Bsu = BsP[buf];
#pragma unroll
        for (int s = 0; s < 2; ++s) {
            AFrag af[4], bfr[2];
#pragma unroll
            for (int mt = 0; mt < 4; ++mt) {
                int row  = wm * 64 + mt * 16 + l16;
                int base = row * (BK / 2) + s * 16 + hs * 4;
                *(uint4*)&af[mt].u[0] = *(const uint4*)&Asu[base];
                *(uint4*)&af[mt].u[4] = *(const uint4*)&Asu[base + 8];
            }
#pragma unroll
            for (int nt = 0; nt < 2; ++nt) {
                int col = wn * 32 + nt * 16 + l16;
                int bb  = col * BPAD + s * 16 + hs * 4;
                *(uint4*)&bfr[nt].u[0] = *(const uint4*)&Bsu[bb];
                *(uint4*)&bfr[nt].u[4] = *(const uint4*)&Bsu[bb + 8];
            }
#pragma unroll
            for (int mt = 0; mt < 4; ++mt)
#pragma unroll
                for (int nt = 0; nt < 2; ++nt)
                    acc[mt][nt] = __builtin_amdgcn_wmma_f32_16x16x32_bf16(
                        false, af[mt].v, false, bfr[nt].v,
                        (short)0, acc[mt][nt], false, false);
        }
        __syncthreads();
    }

    // epilogue (C layout: row = v + 8*hs, col = l16)
#pragma unroll
    for (int mt = 0; mt < 4; ++mt) {
#pragma unroll
        for (int nt = 0; nt < 2; ++nt) {
            int col = bn0 + wn * 32 + nt * 16 + l16;
            float b = BIAS_ ? bias[col] : 0.0f;
#pragma unroll
            for (int v = 0; v < 8; ++v) {
                int row = bm0 + wm * 64 + mt * 16 + v + (hs ? 8 : 0);
                float x = acc[mt][nt][v] + b;
                if (GELU_) x = 0.5f * x * (1.0f + erff(x * 0.70710678118654752f));
                if (RES_)  x += residual[(size_t)row * N + col];
                size_t o = (size_t)row * N + col;
                if (OUTF_) outF[o] = x;
                if (OUTB_) outB[o] = f2bf(x);
            }
        }
    }
}

// ---------------------------------------------------------------------------
// Flash attention. QKV bf16 [NS,2304] (Q|K|V, head cols a*64+d). Z bf16 [NS,768].
// Grid (S/128, A, N), block 256. Block-cooperative K/V staging per 32-key chunk;
// each wave owns 16 query rows and runs online softmax in the C-fragment layout.
// ---------------------------------------------------------------------------
__global__ __launch_bounds__(256)
void attn_wmma(const unsigned short* __restrict__ QKV,
               unsigned short* __restrict__ Z,
               const long long* __restrict__ mask) {
    __shared__ unsigned int   KsP[32 * 36];      // [key][dimpair], pad 36
    __shared__ unsigned int   VsP[64 * 20];      // [dim][keypair], pad 20
    __shared__ unsigned short Plds[8][16 * 32];  // per-wave P transpose

    const int tid  = threadIdx.x;
    const int lane = tid & 31;
    const int wid  = tid >> 5;
    const int l16  = lane & 15;
    const int hs   = (lane >> 4) & 1;

    const int n  = blockIdx.z;
    const int a  = blockIdx.y;
    const int q0 = blockIdx.x * 128 + wid * 16;

    const float maskbias = (mask[n] == 0) ? -__builtin_inff() : 0.0f;

    // Q fragments (16 rows x 64 dims -> 2 K-steps of 32), A-fragment layout
    AFrag qf[2];
    {
        size_t qrow = (size_t)(n * 1024 + q0 + l16) * 2304 + a * 64;
#pragma unroll
        for (int s = 0; s < 2; ++s)
#pragma unroll
            for (int v = 0; v < 8; ++v) {
                int d = s * 32 + (v < 4 ? 2 * v : 16 + 2 * (v - 4)) + (hs ? 8 : 0);
                qf[s].u[v] = *(const unsigned int*)&QKV[qrow + d];
            }
    }

    float m[8], lsum[8];
    floatx8 o[4];
#pragma unroll
    for (int v = 0; v < 8; ++v) { m[v] = -__builtin_inff(); lsum[v] = 0.0f; }
#pragma unroll
    for (int t = 0; t < 4; ++t) o[t] = zero8();

    for (int kb = 0; kb < 1024; kb += 32) {
        // ---- cooperative staging of K (pair-packed) and V (transposed) ----
#pragma unroll
        for (int i = tid, it = 0; it < 4; i += 256, ++it) {
            int key = i >> 5, dp = i & 31;
            size_t base = (size_t)(n * 1024 + kb + key) * 2304 + a * 64 + 2 * dp;
            KsP[key * 36 + dp] = *(const unsigned int*)&QKV[base + 768];
            unsigned int w = *(const unsigned int*)&QKV[base + 1536];
            ((unsigned short*)VsP)[(2 * dp)     * 40 + key] = (unsigned short)(w & 0xffffu);
            ((unsigned short*)VsP)[(2 * dp + 1) * 40 + key] = (unsigned short)(w >> 16);
        }
        if (kb + 32 < 1024)
            __builtin_prefetch(&QKV[(size_t)(n * 1024 + kb + 32 + (tid & 31)) * 2304 + 768 + a * 64], 0, 1);
        __syncthreads();

        // ---- S tile = Q(16x64) * K^T(64x32) ----
        floatx8 c[2] = { zero8(), zero8() };
#pragma unroll
        for (int t = 0; t < 2; ++t) {
#pragma unroll
            for (int s = 0; s < 2; ++s) {
                AFrag kf;
                int bb = (t * 16 + l16) * 36 + s * 16 + hs * 4;
                *(uint4*)&kf.u[0] = *(const uint4*)&KsP[bb];
                *(uint4*)&kf.u[4] = *(const uint4*)&KsP[bb + 8];
                c[t] = __builtin_amdgcn_wmma_f32_16x16x32_bf16(
                    false, qf[s].v, false, kf.v, (short)0, c[t], false, false);
            }
        }
        // ---- online softmax (row stats live in C-layout slot v) ----
#pragma unroll
        for (int v = 0; v < 8; ++v) {
            float a0 = c[0][v] * 0.125f + maskbias;
            float a1 = c[1][v] * 0.125f + maskbias;
            c[0][v] = a0; c[1][v] = a1;
            float mx = fmaxf(a0, a1);
#pragma unroll
            for (int off = 1; off < 16; off <<= 1)
                mx = fmaxf(mx, __shfl_xor(mx, off, 32));
            float mnew = fmaxf(m[v], mx);
            float scale = (mnew == -__builtin_inff()) ? 1.0f : __expf(m[v] - mnew);
            m[v] = mnew;
            float p0 = __expf(c[0][v] - mnew);
            float p1 = __expf(c[1][v] - mnew);
            c[0][v] = p0; c[1][v] = p1;
            float ps = p0 + p1;
#pragma unroll
            for (int off = 1; off < 16; off <<= 1)
                ps += __shfl_xor(ps, off, 32);
            lsum[v] = lsum[v] * scale + ps;
#pragma unroll
            for (int t = 0; t < 4; ++t) o[t][v] *= scale;
        }
        // ---- transpose P (C layout -> A layout) in per-wave LDS region ----
        // LDS ops from one wave complete in order: no block barrier needed.
#pragma unroll
        for (int v = 0; v < 8; ++v) {
            int row = v + (hs ? 8 : 0);
            Plds[wid][row * 32 + l16]      = f2bf(c[0][v]);
            Plds[wid][row * 32 + 16 + l16] = f2bf(c[1][v]);
        }
        AFrag pf;
#pragma unroll
        for (int v = 0; v < 8; ++v) {
            int kk = (v < 4 ? 2 * v : 16 + 2 * (v - 4)) + (hs ? 8 : 0);
            pf.u[v] = *(const unsigned int*)&Plds[wid][l16 * 32 + kk];
        }
        // ---- O += P(16x32) * V(32x64) ----
#pragma unroll
        for (int t = 0; t < 4; ++t) {
            AFrag vf;
            int bb = (t * 16 + l16) * 20 + hs * 4;
            *(uint4*)&vf.u[0] = *(const uint4*)&VsP[bb];
            *(uint4*)&vf.u[4] = *(const uint4*)&VsP[bb + 8];
            o[t] = __builtin_amdgcn_wmma_f32_16x16x32_bf16(
                false, pf.v, false, vf.v, (short)0, o[t], false, false);
        }
        __syncthreads();   // protect KsP/VsP before next chunk restages
    }

    // normalize and store Z (concat heads)
#pragma unroll
    for (int t = 0; t < 4; ++t) {
        int d = t * 16 + l16;
#pragma unroll
        for (int v = 0; v < 8; ++v) {
            int row = q0 + v + (hs ? 8 : 0);
            float val = o[t][v] / lsum[v];
            Z[(size_t)(n * 1024 + row) * 768 + a * 64 + d] = f2bf(val);
        }
    }
}

// ---------------------------------------------------------------------------
// Row LayerNorm over H=768: f32 out (+ optional bf16 copy for next GEMM)
// ---------------------------------------------------------------------------
template<bool OUTB_>
__global__ __launch_bounds__(256)
void layernorm_row(const float* __restrict__ in,
                   const float* __restrict__ gamma,
                   const float* __restrict__ beta,
                   float* __restrict__ outF,
                   unsigned short* __restrict__ outB) {
    const int row  = blockIdx.x;
    const int tid  = threadIdx.x;
    const int lane = tid & 31;
    const int wid  = tid >> 5;
    const float* x = in + (size_t)row * 768;

    float s = 0.f, s2 = 0.f;
#pragma unroll
    for (int i = tid, it = 0; it < 3; i += 256, ++it) {
        float v = x[i];
        s += v; s2 += v * v;
    }
#pragma unroll
    for (int off = 16; off >= 1; off >>= 1) {
        s  += __shfl_xor(s,  off, 32);
        s2 += __shfl_xor(s2, off, 32);
    }
    __shared__ float ws1[8], ws2[8];
    if (lane == 0) { ws1[wid] = s; ws2[wid] = s2; }
    __syncthreads();
    float ts = 0.f, ts2 = 0.f;
#pragma unroll
    for (int i = 0; i < 8; ++i) { ts += ws1[i]; ts2 += ws2[i]; }
    float mu  = ts * (1.0f / 768.0f);
    float var = ts2 * (1.0f / 768.0f) - mu * mu;
    float inv = rsqrtf(var + 1e-5f);
#pragma unroll
    for (int i = tid, it = 0; it < 3; i += 256, ++it) {
        float y = (x[i] - mu) * inv * gamma[i] + beta[i];
        size_t o = (size_t)row * 768 + i;
        outF[o] = y;
        if (OUTB_) outB[o] = f2bf(y);
    }
}

// ---------------------------------------------------------------------------
// Host orchestration
// ---------------------------------------------------------------------------
extern "C" void kernel_launch(void* const* d_in, const int* in_sizes, int n_in,
                              void* d_out, int out_size, void* d_ws, size_t ws_size,
                              hipStream_t stream) {
    (void)in_sizes; (void)n_in; (void)out_size; (void)ws_size;

    const float* X     = (const float*)d_in[0];
    const float* WQ    = (const float*)d_in[1];
    const float* WK    = (const float*)d_in[2];
    const float* WV    = (const float*)d_in[3];
    const float* WO    = (const float*)d_in[4];
    const float* W1    = (const float*)d_in[5];
    const float* b1    = (const float*)d_in[6];
    const float* W2    = (const float*)d_in[7];
    const float* b2    = (const float*)d_in[8];
    const float* gamma = (const float*)d_in[9];
    const float* beta  = (const float*)d_in[10];
    const long long* mask = (const long long*)d_in[11];
    float* out = (float*)d_out;

    const int NS = 10240, H = 768, F = 3072, QN = 2304;

    char* ws = (char*)d_ws;
    size_t off = 0;
    auto alloc = [&](size_t bytes) { char* p = ws + off; off += (bytes + 255) & ~(size_t)255; return p; };
    unsigned short* Xb   = (unsigned short*)alloc((size_t)NS * H * 2);
    unsigned int*   Wqkv = (unsigned int*)  alloc((size_t)(H / 2) * QN * 4);
    unsigned int*   WOb  = (unsigned int*)  alloc((size_t)(H / 2) * H * 4);
    unsigned int*   W1b  = (unsigned int*)  alloc((size_t)(H / 2) * F * 4);
    unsigned int*   W2b  = (unsigned int*)  alloc((size_t)(F / 2) * H * 4);
    unsigned short* QKV  = (unsigned short*)alloc((size_t)NS * QN * 2);
    unsigned short* Zb   = (unsigned short*)alloc((size_t)NS * H * 2);
    float*          O    = (float*)         alloc((size_t)NS * H * 4);
    float*          X1   = (float*)         alloc((size_t)NS * H * 4);
    unsigned short* X1b  = (unsigned short*)alloc((size_t)NS * H * 2);
    unsigned short* G    = (unsigned short*)alloc((size_t)NS * F * 2);
    float*          F2   = (float*)         alloc((size_t)NS * H * 4);

    // 1) convert / pack
    cvt_bf16<<<(NS * H + 255) / 256, 256, 0, stream>>>(X, Xb, NS * H);
    pack_qkv_pairs<<<(384 * QN + 255) / 256, 256, 0, stream>>>(WQ, WK, WV, Wqkv);
    cvt_pack_pairs<<<((H / 2) * H + 255) / 256, 256, 0, stream>>>(WO, WOb, H, H);
    cvt_pack_pairs<<<((H / 2) * F + 255) / 256, 256, 0, stream>>>(W1, W1b, H, F);
    cvt_pack_pairs<<<((F / 2) * H + 255) / 256, 256, 0, stream>>>(W2, W2b, F, H);

    // 2) fused QKV projection: [NS,768] x [768,2304] -> bf16
    gemm_bf16_wmma<false,false,false,false,true><<<dim3(QN / BN, NS / BM), 256, 0, stream>>>(
        Xb, Wqkv, nullptr, nullptr, nullptr, QKV, NS, QN, H);

    // 3) flash attention -> Zb bf16 [NS,768]
    attn_wmma<<<dim3(1024 / 128, 12, 10), 256, 0, stream>>>(QKV, Zb, mask);

    // 4) O projection + residual X -> f32, then LN -> X1 (f32 + bf16)
    gemm_bf16_wmma<false,false,true,true,false><<<dim3(H / BN, NS / BM), 256, 0, stream>>>(
        Zb, WOb, nullptr, X, O, nullptr, NS, H, H);
    layernorm_row<true><<<NS, 256, 0, stream>>>(O, gamma, beta, X1, X1b);

    // 5) FFN: gelu(X1*W1+b1)*W2 + b2 + X1, then LN -> out
    gemm_bf16_wmma<true,true,false,false,true><<<dim3(F / BN, NS / BM), 256, 0, stream>>>(
        X1b, W1b, b1, nullptr, nullptr, G, NS, F, H);
    gemm_bf16_wmma<true,false,true,true,false><<<dim3(H / BN, NS / BM), 256, 0, stream>>>(
        G, W2b, b2, X1, F2, nullptr, NS, H, F);
    layernorm_row<false><<<NS, 256, 0, stream>>>(F2, gamma, beta, out, nullptr);
}
